// APPNP_26225070309442
// MI455X (gfx1250) — compile-verified, matrix operand order checked by reading, not run on hardware
//
#include <hip/hip_runtime.h>
#include <hip/hip_bf16.h>

// ---------------------------------------------------------------------------
// APPNP: 3-layer MLP (bf16 WMMA GEMMs, f32 accumulate, TDM async LDS staging)
// + 10x edge propagation + log_softmax.  gfx1250 (CDNA5, wave32, WMMA, TDM).
// ---------------------------------------------------------------------------

typedef __attribute__((ext_vector_type(16))) __bf16 bf16x16;
typedef __attribute__((ext_vector_type(8)))  float  floatx8;
typedef unsigned int u32x4 __attribute__((ext_vector_type(4)));
typedef int          i32x8 __attribute__((ext_vector_type(8)));
typedef int          i32x4 __attribute__((ext_vector_type(4)));

#define IN_C   512
#define HID_C  256
#define OUT_C  64
#define K_ITERS 10
#define ALPHA  0.1f

// ---- helpers --------------------------------------------------------------

__device__ __forceinline__ unsigned short f2bf(float f) {
    unsigned int u = __float_as_uint(f);
    u += 0x7FFFu + ((u >> 16) & 1u);   // round-to-nearest-even
    return (unsigned short)(u >> 16);
}

__device__ __forceinline__ void atomicAddF(float* p, float v) {
    __hip_atomic_fetch_add(p, v, __ATOMIC_RELAXED, __HIP_MEMORY_SCOPE_AGENT);
}

// Generic pointer to a __shared__ object: low 32 bits are the LDS byte offset
// (flat aperture maps LDS via addr[31:0]).
__device__ __forceinline__ unsigned lds_off(const void* p) {
    return (unsigned)(unsigned long long)p;
}

// ---- Tensor Data Mover: async 2D bf16 tile  global -> LDS -----------------
// D# per CDNA5 ISA 8.3/8.4: group0 = {flags/count, lds_addr, global_addr, type},
// group1 = {pad cfg + data_size, dims, tile dims, stride}.  Rows are tile0
// elements (64B) each followed by 16B of LDS padding -> 80B pitch
// (pad_interval=3 -> 16 DWORDs, pad_amount=3 -> 4 DWORDs).  Rows/cols past
// tensor_dim* are zero-filled by the TDM (handles the partial last M tile).
__device__ __forceinline__ void tdm_load_2d_bf16(
    unsigned ldsByteOff, const void* gptr,
    unsigned tdim0, unsigned tdim1,        // remaining tensor extent (elements)
    unsigned tile0, unsigned tile1,        // tile size (elements)
    unsigned stride0)                      // row stride (elements)
{
    unsigned long long ga = (unsigned long long)gptr;
    u32x4 g0 = {
        1u,                                            // count=1 (valid, user)
        ldsByteOff,                                    // lds_addr
        (unsigned)(ga & 0xFFFFFFFFu),                  // global_addr[31:0]
        (unsigned)((ga >> 32) & 0x01FFFFFFu) | 0x80000000u  // [56:32] | type=2
    };
    i32x8 g1 = {
        (int)((1u << 16) |                 // data_size = 1 -> 2 bytes
              (1u << 20) |                 // pad_enable
              (3u << 22) |                 // pad_interval: 16 DWORDs (64B)
              (3u << 25)),                 // pad_amount:   4 DWORDs (16B)
        (int)((tdim0 & 0xFFFFu) << 16),                         // tensor_dim0 lo
        (int)((tdim0 >> 16) | ((tdim1 & 0xFFFFu) << 16)),       // dim0 hi|dim1 lo
        (int)((tdim1 >> 16) | (tile0 << 16)),                   // dim1 hi|tile0
        (int)(tile1 & 0xFFFFu),                                 // tile1 (tile2=0)
        (int)stride0,                                           // stride0 lo32
        0, 0
    };
    i32x4 z4 = {0, 0, 0, 0};
#if defined(__clang_major__) && (__clang_major__ >= 23)
    i32x8 z8 = {0, 0, 0, 0, 0, 0, 0, 0};
    __builtin_amdgcn_tensor_load_to_lds(g0, g1, z4, z4, z8, 0);
#else
    __builtin_amdgcn_tensor_load_to_lds(g0, g1, z4, z4, 0);
#endif
}

// Load one 16x32 bf16 fragment (CDNA5 ISA 16-bit A-matrix layout) from an LDS
// row of 32 bf16 values (row pitch ROWP halves).  lane = lk*16 + lm; VGPR v
// holds k = (v&3)*2 + (v>>2)*16 + lk*8 (pairs contiguous -> 2x ds_load_b128).
template <int ROWP>
__device__ __forceinline__ bf16x16 load_frag(const unsigned short* rowp, int lk) {
    bf16x16 f;
#pragma unroll
    for (int v = 0; v < 8; ++v) {
        int k0 = ((v & 3) << 1) + ((v >> 2) << 4) + (lk << 3);
        union { unsigned int u; __bf16 h[2]; } c;
        c.u = *(const unsigned int*)(rowp + k0);
        f[2 * v]     = c.h[0];
        f[2 * v + 1] = c.h[1];
    }
    return f;
}

// ---- WMMA GEMM: C[M,Nt] = act(A[M,Kt](bf16) @ BT[Nt,Kt](bf16)^T + bias) ---
// Block = 256 threads (8 waves), tile = 128(M) x 64(N), K in 32-wide chunks.
// TDM double-buffers both LDS tiles: wave 0 issues the DMA for chunk c+1,
// all waves run 4 WMMAs on chunk c, then s_wait_tensorcnt + barrier.

template <int Nt, int Kt, bool RELU, bool OUT_BF16>
__global__ __launch_bounds__(256) void gemm_wmma_kernel(
    const unsigned short* __restrict__ A, const unsigned short* __restrict__ BT,
    const float* __restrict__ bias, void* __restrict__ Cout, int M)
{
    constexpr int ROWP   = 40;        // 80B row pitch: conflict-free b128 frags
    constexpr int NCHUNK = Kt / 32;

    __shared__ unsigned short As[2][128][ROWP];
    __shared__ unsigned short Bs[2][64][ROWP];

    const int tid  = threadIdx.x;
    const int m0   = blockIdx.y * 128;
    const int n0   = blockIdx.x * 64;
    const int w    = tid >> 5;
    const int lane = tid & 31;
    const int lm   = lane & 15;
    const int lk   = lane >> 4;

    const unsigned aOff[2] = { lds_off(&As[0][0][0]), lds_off(&As[1][0][0]) };
    const unsigned bOff[2] = { lds_off(&Bs[0][0][0]), lds_off(&Bs[1][0][0]) };

    auto issue = [&](int c, int buf) {
        const int kc = c * 32;
        tdm_load_2d_bf16(aOff[buf], A + (size_t)m0 * Kt + kc,
                         (unsigned)(Kt - kc), (unsigned)(M - m0),
                         32u, 128u, (unsigned)Kt);
        tdm_load_2d_bf16(bOff[buf], BT + (size_t)n0 * Kt + kc,
                         (unsigned)(Kt - kc), (unsigned)(Nt - n0),
                         32u, 64u, (unsigned)Kt);
    };

    floatx8 acc[4] = {};

    if (w == 0) {
        issue(0, 0);
        __builtin_amdgcn_s_wait_tensorcnt(0);
    }
    __syncthreads();

    for (int c = 0; c < NCHUNK; ++c) {
        const int cur = c & 1;
        if (w == 0 && c + 1 < NCHUNK)
            issue(c + 1, cur ^ 1);         // async DMA overlaps WMMAs below

        bf16x16 af = load_frag<ROWP>(&As[cur][w * 16 + lm][0], lk);
        bf16x16 bfr[4];
#pragma unroll
        for (int j = 0; j < 4; ++j)
            bfr[j] = load_frag<ROWP>(&Bs[cur][j * 16 + lm][0], lk);
#pragma unroll
        for (int j = 0; j < 4; ++j)
            acc[j] = __builtin_amdgcn_wmma_f32_16x16x32_bf16(
                false, af, false, bfr[j], (short)0, acc[j], false, false);

        if (c + 1 < NCHUNK) {
            if (w == 0) __builtin_amdgcn_s_wait_tensorcnt(0);
            __syncthreads();
        }
    }

    // epilogue: bias (+ReLU).  C/D layout: lane group lk -> M = w*16 + lk*8 + r
#pragma unroll
    for (int j = 0; j < 4; ++j) {
        int n = n0 + j * 16 + lm;
        float b = bias[n];
#pragma unroll
        for (int r = 0; r < 8; ++r) {
            int m = m0 + w * 16 + lk * 8 + r;
            if (m < M) {
                float v = acc[j][r] + b;
                if (RELU) v = fmaxf(v, 0.f);
                if (OUT_BF16)
                    ((unsigned short*)Cout)[(size_t)m * Nt + n] = f2bf(v);
                else
                    ((float*)Cout)[(size_t)m * Nt + n] = v;
            }
        }
    }
}

// ---- small utility kernels ------------------------------------------------

// vectorized f32 -> bf16 (n must be a multiple of 4; all our sizes are)
__global__ void k_f32_to_bf16(const float* __restrict__ src,
                              unsigned short* __restrict__ dst, int n4) {
    int i = blockIdx.x * blockDim.x + threadIdx.x;
    if (i < n4) {
        float4 v = ((const float4*)src)[i];
        uint2 o;
        o.x = (unsigned int)f2bf(v.x) | ((unsigned int)f2bf(v.y) << 16);
        o.y = (unsigned int)f2bf(v.z) | ((unsigned int)f2bf(v.w) << 16);
        ((uint2*)dst)[i] = o;
    }
}

// weights: f32 [K,N] -> bf16 transposed [N,K] (B tiles become row-major 2D)
__global__ void k_w_to_bf16_t(const float* __restrict__ src,
                              unsigned short* __restrict__ dst, int K, int N) {
    int i = blockIdx.x * blockDim.x + threadIdx.x;
    if (i < K * N) {
        int k = i / N;
        int n = i - k * N;
        dst[(size_t)n * K + k] = f2bf(src[i]);
    }
}

__global__ void k_zero(float* __restrict__ p, int n) {
    int i = blockIdx.x * blockDim.x + threadIdx.x;
    if (i < n) p[i] = 0.f;
}

__global__ void k_copy(const float* __restrict__ a, float* __restrict__ b, int n) {
    int i = blockIdx.x * blockDim.x + threadIdx.x;
    if (i < n) b[i] = a[i];
}

__global__ void k_degree(const long long* __restrict__ rows,
                         float* __restrict__ deg, int E) {
    int e = blockIdx.x * blockDim.x + threadIdx.x;
    if (e < E) atomicAddF(&deg[rows[e]], 1.0f);
}

__global__ void k_dinv(float* __restrict__ deg, int n) {
    int i = blockIdx.x * blockDim.x + threadIdx.x;
    if (i < n) {
        float d = deg[i];
        deg[i] = (d > 0.f) ? rsqrtf(fmaxf(d, 1.0f)) : 0.f;
    }
}

__global__ void k_norm(const long long* __restrict__ rows,
                       const long long* __restrict__ cols,
                       const float* __restrict__ dinv,
                       float* __restrict__ norm, int E) {
    int e = blockIdx.x * blockDim.x + threadIdx.x;
    if (e < E) norm[e] = dinv[rows[e]] * dinv[cols[e]];
}

// 4 edges per 256-thread block; 64 channels per edge (coalesced gather+scatter)
__global__ void k_scatter(const long long* __restrict__ rows,
                          const long long* __restrict__ cols,
                          const float* __restrict__ norm,
                          const float* __restrict__ z,
                          float* __restrict__ agg, int E) {
    int t = blockIdx.x * blockDim.x + threadIdx.x;
    int e = t >> 6;
    int c = t & 63;
    if (e >= E) return;
    long long r  = rows[e];
    long long cl = cols[e];
    float v = z[(size_t)cl * OUT_C + c] * norm[e];
    atomicAddF(&agg[(size_t)r * OUT_C + c], v);
}

__global__ void k_update(const float* __restrict__ agg,
                         const float* __restrict__ h,
                         float* __restrict__ z, int n) {
    int i = blockIdx.x * blockDim.x + threadIdx.x;
    if (i < n) z[i] = (1.0f - ALPHA) * agg[i] + ALPHA * h[i];
}

__global__ void k_log_softmax(const float* __restrict__ z,
                              float* __restrict__ out, int nRows) {
    int row = blockIdx.x;
    if (row >= nRows) return;
    int t = threadIdx.x;                 // 64 threads, one channel each
    __shared__ float red[64];
    float v = z[(size_t)row * OUT_C + t];
    red[t] = v;
    __syncthreads();
#pragma unroll
    for (int s = 32; s > 0; s >>= 1) {
        if (t < s) red[t] = fmaxf(red[t], red[t + s]);
        __syncthreads();
    }
    float mx = red[0];
    __syncthreads();
    float e = expf(v - mx);
    red[t] = e;
    __syncthreads();
#pragma unroll
    for (int s = 32; s > 0; s >>= 1) {
        if (t < s) red[t] += red[t + s];
        __syncthreads();
    }
    float lse = logf(red[0]);
    out[(size_t)row * OUT_C + t] = v - mx - lse;
}

// ---- launch ---------------------------------------------------------------

extern "C" void kernel_launch(void* const* d_in, const int* in_sizes, int n_in,
                              void* d_out, int out_size, void* d_ws, size_t ws_size,
                              hipStream_t stream) {
    const float*     x  = (const float*)d_in[0];
    const long long* ei = (const long long*)d_in[1];   // [2, E] int64
    const float* W1 = (const float*)d_in[2];
    const float* b1 = (const float*)d_in[3];
    const float* W2 = (const float*)d_in[4];
    const float* b2 = (const float*)d_in[5];
    const float* W3 = (const float*)d_in[6];
    const float* b3 = (const float*)d_in[7];
    float* out = (float*)d_out;

    const int Nn = in_sizes[0] / IN_C;     // 100000 nodes
    const int E  = in_sizes[1] / 2;        // 1.6M edges
    const long long* rows = ei;
    const long long* cols = ei + E;

    // carve workspace (256B aligned)
    char* p = (char*)d_ws;
    auto take = [&](size_t bytes) -> void* {
        void* q = (void*)p;
        p += (bytes + 255) & ~(size_t)255;
        return q;
    };
    unsigned short* xb   = (unsigned short*)take((size_t)Nn * IN_C * 2);
    unsigned short* Wt1  = (unsigned short*)take((size_t)IN_C  * HID_C * 2);
    unsigned short* Wt2  = (unsigned short*)take((size_t)HID_C * HID_C * 2);
    unsigned short* Wt3  = (unsigned short*)take((size_t)HID_C * OUT_C * 2);
    unsigned short* h1   = (unsigned short*)take((size_t)Nn * HID_C * 2);
    unsigned short* h2   = (unsigned short*)take((size_t)Nn * HID_C * 2);
    float* h    = (float*)take((size_t)Nn * OUT_C * 4);
    float* z    = (float*)take((size_t)Nn * OUT_C * 4);
    float* agg  = (float*)take((size_t)Nn * OUT_C * 4);
    float* dinv = (float*)take((size_t)Nn * 4);            // deg then dinv in-place
    float* norm = (float*)take((size_t)E * 4);

    // 1) x -> bf16 (row-major); weights -> bf16 transposed [N,K]
    {
        int n4 = (Nn * IN_C) / 4;
        k_f32_to_bf16<<<(n4 + 255) / 256, 256, 0, stream>>>(x, xb, n4);
        int n1 = IN_C * HID_C, n2 = HID_C * HID_C, n3 = HID_C * OUT_C;
        k_w_to_bf16_t<<<(n1 + 255) / 256, 256, 0, stream>>>(W1, Wt1, IN_C,  HID_C);
        k_w_to_bf16_t<<<(n2 + 255) / 256, 256, 0, stream>>>(W2, Wt2, HID_C, HID_C);
        k_w_to_bf16_t<<<(n3 + 255) / 256, 256, 0, stream>>>(W3, Wt3, HID_C, OUT_C);
    }

    // 2) MLP via WMMA + TDM double-buffered staging
    dim3 gHid(HID_C / 64, (Nn + 127) / 128);
    dim3 gOut(OUT_C / 64, (Nn + 127) / 128);
    gemm_wmma_kernel<HID_C, IN_C,  true,  true ><<<gHid, 256, 0, stream>>>(xb, Wt1, b1, h1, Nn);
    gemm_wmma_kernel<HID_C, HID_C, true,  true ><<<gHid, 256, 0, stream>>>(h1, Wt2, b2, h2, Nn);
    gemm_wmma_kernel<OUT_C, HID_C, false, false><<<gOut, 256, 0, stream>>>(h2, Wt3, b3, h,  Nn);

    // 3) edge normalization: deg -> dinv -> norm
    k_zero  <<<(Nn + 255) / 256, 256, 0, stream>>>(dinv, Nn);
    k_degree<<<(E + 255) / 256, 256, 0, stream>>>(rows, dinv, E);
    k_dinv  <<<(Nn + 255) / 256, 256, 0, stream>>>(dinv, Nn);
    k_norm  <<<(E + 255) / 256, 256, 0, stream>>>(rows, cols, dinv, norm, E);

    // 4) APPNP propagation: z = h; 10x { agg = scatter(z*norm); z = 0.9*agg + 0.1*h }
    const int nZ = Nn * OUT_C;
    k_copy<<<(nZ + 255) / 256, 256, 0, stream>>>(h, z, nZ);
    for (int it = 0; it < K_ITERS; ++it) {
        k_zero   <<<(nZ + 255) / 256, 256, 0, stream>>>(agg, nZ);
        k_scatter<<<(E + 3) / 4, 256, 0, stream>>>(rows, cols, norm, z, agg, E);
        k_update <<<(nZ + 255) / 256, 256, 0, stream>>>(agg, h, z, nZ);
    }

    // 5) log_softmax rows of 64
    k_log_softmax<<<Nn, 64, 0, stream>>>(z, out, Nn);
}
